// Quantize_20882130993469
// MI455X (gfx1250) — matrix operand order repaired and checked by reading, not compiled, not run on hardware
//
#include <hip/hip_runtime.h>
#include <stdint.h>

// ---------------------------------------------------------------- types ----
typedef __attribute__((ext_vector_type(16))) __bf16   v16bf;
typedef __attribute__((ext_vector_type(8)))  float    v8f;
typedef __attribute__((ext_vector_type(4)))  uint32_t u32x4;
typedef __attribute__((ext_vector_type(4)))  int      v4i;

// ------------------------------------------------------------- problem -----
#define T_DIM   2048
#define B_DIM   8
#define N_ROWS  (T_DIM * B_DIM)     // 16384
#define D_DIM   512
#define K_DIM   4096
#define N_ITILE (N_ROWS / 16)       // 1024 row tiles (M)
#define N_JTILE (K_DIM / 16)        // 256 code tiles (N)
#define N_DSB   (D_DIM / 32)        // 16 K-steps of 32
#define N_JCHUNK 4
#define JT_PER_CHUNK (N_JTILE / N_JCHUNK)
#define WAVES_PER_BLOCK 4

// output layout (floats), concatenated in reference return order
#define OUT_Q    ((size_t)0)
#define OUT_DIFF ((size_t)(N_ROWS) * D_DIM)          // 8388608
#define OUT_IND  (OUT_DIFF + 1)                      // 8388609
#define OUT_EFF  (OUT_IND + N_ROWS)                  // 8404993

// B tile in LDS: 16 dsb * 256 dwords (hi) + same (lo) = 8192 dwords = 32 KB
#define BUF_DW   8192

// ------------------------------------------------- async-copy capability ---
#if __has_builtin(__builtin_amdgcn_global_load_async_to_lds_b128)
#define USE_ASYNC 1
#define GPTR(p) ((__attribute__((address_space(1))) v4i*)(p))
#define LPTR(p) ((__attribute__((address_space(3))) v4i*)(p))
#if __has_builtin(__builtin_amdgcn_s_wait_asynccnt)
#define WAIT_ASYNC(n) __builtin_amdgcn_s_wait_asynccnt(n)
#else
#define WAIT_ASYNC(n) asm volatile("s_wait_asynccnt %0" ::"n"(n) : "memory")
#endif
#else
#define USE_ASYNC 0
#endif

// ---------------------------------------------------------- bf16 helpers ---
__device__ __forceinline__ uint32_t f2bf(float x) {
  uint32_t u = __float_as_uint(x);
  return (u + 0x7FFFu + ((u >> 16) & 1u)) >> 16;    // round-to-nearest-even
}
__device__ __forceinline__ float bf2f(uint32_t h) { return __uint_as_float(h << 16); }

// ---------------------------------------------------------------------------
// Swizzle X (f32 [N_ROWS, D]) into A-fragment layout, split hi/lo bf16.
// dword index t = ((itile*16 + dsb)*32 + lane)*8 + v
// A 16x32 bf16 layout: lane = half*16 + m; VGPR v holds K = (v&4?16:0)+(v&3)*2+half*8 (+1)
__global__ __launch_bounds__(256) void vq_prep_a(const float* __restrict__ X,
                                                 uint32_t* __restrict__ Ahi,
                                                 uint32_t* __restrict__ Alo) {
  int t    = blockIdx.x * blockDim.x + threadIdx.x;   // 0 .. 4194303
  int v    = t & 7;
  int lane = (t >> 3) & 31;
  int blk  = t >> 8;                                  // itile*16 + dsb
  int dsb  = blk & 15;
  int it   = blk >> 4;
  int half = lane >> 4, m = lane & 15;
  int kk   = dsb * 32 + ((v & 4) ? 16 : 0) + (v & 3) * 2 + half * 8;
  int row  = it * 16 + m;
  float x0 = X[row * D_DIM + kk];
  float x1 = X[row * D_DIM + kk + 1];
  uint32_t h0 = f2bf(x0), h1 = f2bf(x1);
  uint32_t l0 = f2bf(x0 - bf2f(h0)), l1 = f2bf(x1 - bf2f(h1));
  Ahi[t] = h0 | (h1 << 16);
  Alo[t] = l0 | (l1 << 16);
}

// Swizzle E (f32 [D, K]) into B-fragment layout, split hi/lo bf16.
// B 32x16 bf16 layout: lane = khalf*16 + n; VGPR v holds K = (v&4?16:0)+(v&3)*2+khalf*8 (+1)
__global__ __launch_bounds__(256) void vq_prep_b(const float* __restrict__ E,
                                                 uint32_t* __restrict__ Bhi,
                                                 uint32_t* __restrict__ Blo) {
  int t     = blockIdx.x * blockDim.x + threadIdx.x;  // 0 .. 1048575
  int v     = t & 7;
  int lane  = (t >> 3) & 31;
  int blk   = t >> 8;                                 // jtile*16 + dsb
  int dsb   = blk & 15;
  int jt    = blk >> 4;
  int khalf = lane >> 4, n = lane & 15;
  int d     = dsb * 32 + ((v & 4) ? 16 : 0) + (v & 3) * 2 + khalf * 8;
  int j     = jt * 16 + n;
  float e0 = E[(size_t)d * K_DIM + j];
  float e1 = E[(size_t)(d + 1) * K_DIM + j];
  uint32_t h0 = f2bf(e0), h1 = f2bf(e1);
  uint32_t l0 = f2bf(e0 - bf2f(h0)), l1 = f2bf(e1 - bf2f(h1));
  Bhi[t] = h0 | (h1 << 16);
  Blo[t] = l0 | (l1 << 16);
}

// halfnorm[j] = 0.5 * sum_d E[d][j]^2 ; also zero the usage histogram.
__global__ __launch_bounds__(256) void vq_prep_norm(const float* __restrict__ E,
                                                    float* __restrict__ halfnorm,
                                                    unsigned int* __restrict__ counts) {
  int j = blockIdx.x * blockDim.x + threadIdx.x;
  if (j < K_DIM) {
    float s = 0.f;
    for (int d = 0; d < D_DIM; ++d) {
      float e = E[(size_t)d * K_DIM + j];
      s += e * e;
    }
    halfnorm[j] = 0.5f * s;
    counts[j]   = 0u;
  }
}

// ---------------------------------------------------------------------------
// Stage one 32 KB swizzled B tile (hi 16KB | lo 16KB) into an LDS buffer.
// Block-wide: 128 threads x 16 B x 16 batches. Async path: ASYNCcnt += 16/wave.
__device__ __forceinline__ void stage_b(const uint32_t* __restrict__ Bhi,
                                        const uint32_t* __restrict__ Blo,
                                        uint32_t* sm, int buf, int jt, int tid) {
  const uint32_t* gh = Bhi + (size_t)jt * 4096 + tid * 4;
  const uint32_t* gl = Blo + (size_t)jt * 4096 + tid * 4;
  uint32_t* lh = sm + buf * BUF_DW + tid * 4;
  uint32_t* ll = lh + 4096;
#if USE_ASYNC
#pragma unroll
  for (int c = 0; c < 8; ++c) {
    __builtin_amdgcn_global_load_async_to_lds_b128(GPTR(gh + c * 512),
                                                   LPTR(lh + c * 512), 0, 0);
    __builtin_amdgcn_global_load_async_to_lds_b128(GPTR(gl + c * 512),
                                                   LPTR(ll + c * 512), 0, 0);
  }
#else
#pragma unroll
  for (int c = 0; c < 8; ++c) {
    u32x4 a = *(const u32x4*)(gh + c * 512);
    u32x4 b = *(const u32x4*)(gl + c * 512);
    *(u32x4*)(lh + c * 512) = a;
    *(u32x4*)(ll + c * 512) = b;
  }
#endif
}

// ---------------------------------------------------------------------------
// Fused GEMM + argmax. 4 waves/block share B tiles through double-buffered LDS
// (4x reduction in B L2 traffic); each wave owns one 16-row tile, A fragments
// are reloaded from global memory for every j tile.
// score[i][j] = x_i . e_j - 0.5||e_j||^2 ; argmax score == argmin dist.
// Split precision: C += Ahi*Bhi + Ahi*Blo + Alo*Bhi (3 bf16 WMMAs / K-step).
__global__ __launch_bounds__(128) void vq_gemm_argmax(
    const uint32_t* __restrict__ Ahi, const uint32_t* __restrict__ Alo,
    const uint32_t* __restrict__ Bhi, const uint32_t* __restrict__ Blo,
    const float* __restrict__ halfnorm, float* __restrict__ best) {
  __shared__ uint32_t smem[2 * BUF_DW];               // 64 KB, double buffer

  const int tid  = threadIdx.x;
  const int wave = tid >> 5;
  const int lane = tid & 31;
  const int it   = blockIdx.x * WAVES_PER_BLOCK + wave;  // 0..1023
  const int jc   = blockIdx.y;                            // 0..3
  const int j0   = jc * JT_PER_CHUNK, j1 = j0 + JT_PER_CHUNK;

  float bv[8];
  int   bi[8];
#pragma unroll
  for (int r = 0; r < 8; ++r) { bv[r] = -__builtin_inff(); bi[r] = 0x7fffffff; }

  const uint32_t* aHiBase = Ahi + (size_t)it * (N_DSB * 256) + lane * 8;
  const uint32_t* aLoBase = Alo + (size_t)it * (N_DSB * 256) + lane * 8;

  stage_b(Bhi, Blo, smem, 0, j0, tid);                // prologue: tile j0

  for (int jt = j0; jt < j1; ++jt) {
    const int buf = (jt - j0) & 1;
    if (jt + 1 < j1) {
      stage_b(Bhi, Blo, smem, buf ^ 1, jt + 1, tid);  // prefetch next tile
#if USE_ASYNC
      WAIT_ASYNC(16);                                 // in-order: tile jt landed
#endif
    } else {
#if USE_ASYNC
      WAIT_ASYNC(0);
#endif
    }
    __syncthreads();                                  // tile jt visible to all waves

    v8f c = {0.f, 0.f, 0.f, 0.f, 0.f, 0.f, 0.f, 0.f};
    const uint32_t* lb = smem + buf * BUF_DW + lane * 8;
#pragma unroll
    for (int dsb = 0; dsb < N_DSB; ++dsb) {
      union { u32x4 q[2]; v16bf v; } ah, al, bh, bl;
      const u32x4* pah = (const u32x4*)(aHiBase + dsb * 256);
      const u32x4* pal = (const u32x4*)(aLoBase + dsb * 256);
      const u32x4* pbh = (const u32x4*)(lb + dsb * 256);
      const u32x4* pbl = (const u32x4*)(lb + 4096 + dsb * 256);
      ah.q[0] = pah[0]; ah.q[1] = pah[1];
      al.q[0] = pal[0]; al.q[1] = pal[1];
      bh.q[0] = pbh[0]; bh.q[1] = pbh[1];
      bl.q[0] = pbl[0]; bl.q[1] = pbl[1];
      c = __builtin_amdgcn_wmma_f32_16x16x32_bf16(false, ah.v, false, bh.v,
                                                  (short)0, c, false, false);
      c = __builtin_amdgcn_wmma_f32_16x16x32_bf16(false, ah.v, false, bl.v,
                                                  (short)0, c, false, false);
      c = __builtin_amdgcn_wmma_f32_16x16x32_bf16(false, al.v, false, bh.v,
                                                  (short)0, c, false, false);
    }
    // C/D layout: VGPR r, lane = (m>=8)*16 + n  ->  element (m = r + 8*half, n)
    const int   n  = lane & 15;
    const int   jj = jt * 16 + n;
    const float hn = halfnorm[jj];
#pragma unroll
    for (int r = 0; r < 8; ++r) {
      float s = c[r] - hn;
      if (s > bv[r]) { bv[r] = s; bi[r] = jj; }   // ascending j: strict > keeps first max
    }
    __syncthreads();                              // all reads done before buf reuse
  }

  // reduce across the 16 n-lanes of each half (xor 1,2,4,8 stays in-half)
#pragma unroll
  for (int off = 1; off <= 8; off <<= 1) {
#pragma unroll
    for (int r = 0; r < 8; ++r) {
      float ov = __shfl_xor(bv[r], off, 32);
      int   oi = __shfl_xor(bi[r], off, 32);
      if (ov > bv[r] || (ov == bv[r] && oi < bi[r])) { bv[r] = ov; bi[r] = oi; }
    }
  }

  if ((lane & 15) == 0) {
    const int half = lane >> 4;
#pragma unroll
    for (int r = 0; r < 8; ++r) {
      int row = it * 16 + half * 8 + r;
      best[((size_t)row * N_JCHUNK + jc) * 2 + 0] = bv[r];
      best[((size_t)row * N_JCHUNK + jc) * 2 + 1] = __int_as_float(bi[r]);
    }
  }
}

// ---------------------------------------------------------------------------
// Per row: combine chunk candidates, gather code vector, write quantize,
// emit index, histogram, and per-row squared-diff partial.
__global__ __launch_bounds__(128) void vq_quantize(
    const float* __restrict__ X, const unsigned char* __restrict__ mask,
    const float* __restrict__ E, const float* __restrict__ best,
    float* __restrict__ qout, float* __restrict__ indout,
    float* __restrict__ rowdiff, unsigned int* __restrict__ counts) {
  const int row = blockIdx.x;
  __shared__ int   sIdx;
  __shared__ float sM;
  __shared__ float sred[128];

  if (threadIdx.x == 0) {
    float bvv = -__builtin_inff();
    int   bii = 0x7fffffff;
    for (int c = 0; c < N_JCHUNK; ++c) {
      float v = best[((size_t)row * N_JCHUNK + c) * 2 + 0];
      int   i = __float_as_int(best[((size_t)row * N_JCHUNK + c) * 2 + 1]);
      if (v > bvv || (v == bvv && i < bii)) { bvv = v; bii = i; }
    }
    float m = mask[row] ? 1.0f : 0.0f;
    sIdx = bii;
    sM   = m;
    indout[row] = (float)bii;
    if (mask[row]) atomicAdd(&counts[bii], 1u);
  }
  __syncthreads();

  const int   idx = sIdx;
  const float m   = sM;
  float ps = 0.f;
  for (int d = threadIdx.x; d < D_DIM; d += 128) {
    float q  = E[(size_t)d * K_DIM + idx] * m;
    float xm = X[(size_t)row * D_DIM + d] * m;
    qout[(size_t)row * D_DIM + d] = xm + (q - xm);   // straight-through expression
    float dd = q - xm;
    ps += dd * dd;
  }
  sred[threadIdx.x] = ps;
  __syncthreads();
  for (int o = 64; o > 0; o >>= 1) {
    if (threadIdx.x < o) sred[threadIdx.x] += sred[threadIdx.x + o];
    __syncthreads();
  }
  if (threadIdx.x == 0) rowdiff[row] = sred[0];
}

// ---------------------------------------------------------------------------
// Single-block, fixed-order reductions: diff and effective_units.
__global__ __launch_bounds__(256) void vq_finalize(
    const float* __restrict__ rowdiff, const unsigned int* __restrict__ counts,
    const unsigned char* __restrict__ mask, float* __restrict__ out) {
  __shared__ double sd[256];
  const int t = threadIdx.x;

  double s = 0.0;
  for (int i = t; i < N_ROWS; i += 256) s += (double)rowdiff[i];
  sd[t] = s; __syncthreads();
  for (int o = 128; o > 0; o >>= 1) { if (t < o) sd[t] += sd[t + o]; __syncthreads(); }
  double diffsum = sd[0]; __syncthreads();

  double c2 = 0.0;
  for (int k = t; k < K_DIM; k += 256) { double c = (double)counts[k]; c2 += c * c; }
  sd[t] = c2; __syncthreads();
  for (int o = 128; o > 0; o >>= 1) { if (t < o) sd[t] += sd[t + o]; __syncthreads(); }
  double sumc2 = sd[0]; __syncthreads();

  double ms = 0.0;
  for (int i = t; i < N_ROWS; i += 256) ms += mask[i] ? 1.0 : 0.0;
  sd[t] = ms; __syncthreads();
  for (int o = 128; o > 0; o >>= 1) { if (t < o) sd[t] += sd[t + o]; __syncthreads(); }
  double msum = sd[0];

  if (t == 0) {
    out[OUT_DIFF] = (float)(diffsum / ((double)N_ROWS * (double)D_DIM));
    out[OUT_EFF]  = (float)((msum * msum) / sumc2);
  }
}

// ---------------------------------------------------------------------------
extern "C" void kernel_launch(void* const* d_in, const int* in_sizes, int n_in,
                              void* d_out, int out_size, void* d_ws, size_t ws_size,
                              hipStream_t stream) {
  (void)in_sizes; (void)n_in; (void)out_size; (void)ws_size;
  const float*         X    = (const float*)d_in[0];          // [T,B,D] f32
  const unsigned char* mask = (const unsigned char*)d_in[1];  // [T,B] bool
  const float*         E    = (const float*)d_in[2];          // [D,K] f32
  float*               out  = (float*)d_out;

  char* ws = (char*)d_ws;
  uint32_t*     Ahi      = (uint32_t*)(ws);                                  // 16 MB
  uint32_t*     Alo      = (uint32_t*)(ws + ((size_t)16 << 20));             // 16 MB
  uint32_t*     Bhi      = (uint32_t*)(ws + ((size_t)32 << 20));             //  4 MB
  uint32_t*     Blo      = (uint32_t*)(ws + ((size_t)36 << 20));             //  4 MB
  float*        halfnorm = (float*)   (ws + ((size_t)40 << 20));             // 16 KB
  unsigned int* counts   = (unsigned int*)(ws + ((size_t)40 << 20) + 16384); // 16 KB
  float*        best     = (float*)   (ws + ((size_t)40 << 20) + 32768);     // 512 KB
  float*        rowdiff  = (float*)   (ws + ((size_t)40 << 20) + 32768 + 524288); // 64 KB

  vq_prep_a<<<(N_ITILE * N_DSB * 32 * 8) / 256, 256, 0, stream>>>(X, Ahi, Alo);
  vq_prep_b<<<(N_JTILE * N_DSB * 32 * 8) / 256, 256, 0, stream>>>(E, Bhi, Blo);
  vq_prep_norm<<<K_DIM / 256, 256, 0, stream>>>(E, halfnorm, counts);
  vq_gemm_argmax<<<dim3(N_ITILE / WAVES_PER_BLOCK, N_JCHUNK), 128, 0, stream>>>(
      Ahi, Alo, Bhi, Blo, halfnorm, best);
  vq_quantize<<<N_ROWS, 128, 0, stream>>>(X, mask, E, best, out + OUT_Q,
                                          out + OUT_IND, rowdiff, counts);
  vq_finalize<<<1, 256, 0, stream>>>(rowdiff, counts, mask, out);
}
